// CHGNetSimple_80917183856999
// MI455X (gfx1250) — matrix operand order repaired and checked by reading, not compiled, not run on hardware
//
#include <hip/hip_runtime.h>
#include <hip/hip_bf16.h>

typedef __bf16 bf16_t;
typedef __attribute__((ext_vector_type(16))) __bf16 v16bf;
typedef __attribute__((ext_vector_type(8)))  float  v8f;
typedef __attribute__((ext_vector_type(2)))  float  v2f;

#define D 64

static __device__ __forceinline__ bf16_t f2bf(float f) {
    unsigned u = __builtin_bit_cast(unsigned, f);
    unsigned r = (u + 0x7FFFu + ((u >> 16) & 1u)) >> 16;
    unsigned short h = (unsigned short)r;
    return __builtin_bit_cast(bf16_t, h);
}

// ---------------------------------------------------------------------------
// Pass 1: gather rows -> [rows,KDIM] bf16 tile in LDS, GEMM vs Wc/Wg (bf16
// WMMA, f32 accum), store pre-activations, accumulate per-column sum/sumsq.
// Block = 256 threads (8 waves): wave -> (matrix = core/gate, col-tile 0..3).
// ---------------------------------------------------------------------------
template<int KDIM>
__global__ __launch_bounds__(256)
void chgnet_gemm_stats(const float* __restrict__ s0, const int* __restrict__ i0,
                       const float* __restrict__ s1, const int* __restrict__ i1,
                       const float* __restrict__ s2, const int* __restrict__ i2,
                       const float* __restrict__ s3,
                       const float* __restrict__ Wc, const float* __restrict__ Wg,
                       float* __restrict__ coreOut, float* __restrict__ gateOut,
                       float* __restrict__ stats, int nrows)
{
    __shared__ bf16_t ldsA[16 * KDIM];

    const int tid  = threadIdx.x;
    const int wave = tid >> 5;
    const int lane = tid & 31;
    const int mat  = wave >> 2;     // 0 = core, 1 = gate
    const int ct   = wave & 3;      // column tile (16 cols each)
    const int half = lane >> 4;
    const int ln   = lane & 15;
    const int col  = ct * 16 + ln;

    // Preload B fragments (weights) into registers: constant across row tiles.
    // B 32x16 bf16, contiguous k order used here (A matches): VGPR v, lanes 0-15: K=2v,2v+1 ; lanes 16-31: K=16+2v,...
    const float* W = mat ? Wg : Wc;
    v16bf bfrag[KDIM / 32];
#pragma unroll
    for (int ks = 0; ks < KDIM / 32; ++ks) {
#pragma unroll
        for (int p = 0; p < 8; ++p) {
            int K = ks * 32 + half * 16 + 2 * p;
            bfrag[ks][2 * p]     = f2bf(W[(K)     * D + col]);
            bfrag[ks][2 * p + 1] = f2bf(W[(K + 1) * D + col]);
        }
    }

    float ssum = 0.f, ssq = 0.f;
    const int ntiles = nrows >> 4;

    for (int tile = blockIdx.x; tile < ntiles; tile += gridDim.x) {
        const int rbase = tile << 4;
        __syncthreads();   // previous iteration's WMMA reads of ldsA are done
        // Cooperative gather of 16 rows x KDIM into LDS as bf16.
        for (int idx = tid; idx < 16 * KDIM; idx += 256) {
            int r   = idx / KDIM;
            int k   = idx - r * KDIM;
            int seg = k >> 6;
            int kk  = k & 63;
            int gr  = rbase + r;
            float v;
            if (seg == 0)      v = s0[(size_t)(i0 ? i0[gr] : gr) * D + kk];
            else if (seg == 1) v = s1[(size_t)(i1 ? i1[gr] : gr) * D + kk];
            else if (seg == 2) v = s2[(size_t)(i2 ? i2[gr] : gr) * D + kk];
            else               v = s3[(size_t)gr * D + kk];
            ldsA[idx] = f2bf(v);
        }
        __syncthreads();

        v8f acc = {};
#pragma unroll
        for (int ks = 0; ks < KDIM / 32; ++ks) {
            // A 16x32 bf16, contiguous k order used here (same as B): VGPR i: K = half*16+i*2
            v16bf a;
#pragma unroll
            for (int i = 0; i < 8; ++i) {
                int grp = i >> 2, j = i & 3;
                int K = ks * 32 + half * 16 + i * 2;
                a[2 * i]     = ldsA[ln * KDIM + K];
                a[2 * i + 1] = ldsA[ln * KDIM + K + 1];
            }
            acc = __builtin_amdgcn_wmma_f32_16x16x32_bf16(
                false, a, false, bfrag[ks], (short)0, acc, false, false);
        }

        float* outp = mat ? gateOut : coreOut;
#pragma unroll
        for (int r = 0; r < 8; ++r) {
            int row = rbase + r + half * 8;      // C layout: VGPR r -> M=r (+8 hi half)
            float v = acc[r];
            outp[(size_t)row * D + col] = v;
            ssum += v;
            ssq  += v * v;
        }
    }

    // stats layout: [coreSum(64), coreSq(64), gateSum(64), gateSq(64)]
    atomicAdd(&stats[(mat * 2 + 0) * D + col], ssum);
    atomicAdd(&stats[(mat * 2 + 1) * D + col], ssq);
}

// ---------------------------------------------------------------------------
// Pass 2: finalize BN, silu(core)*sigmoid(gate); either atomic segment-sum
// scatter (direct==0) or direct write + residual (direct==1, angle stage).
// ---------------------------------------------------------------------------
__global__ __launch_bounds__(256)
void chgnet_bn_gate(const float* __restrict__ core, const float* __restrict__ gate,
                    const float* __restrict__ stats,
                    const float* __restrict__ gc, const float* __restrict__ bc,
                    const float* __restrict__ gg, const float* __restrict__ bg,
                    const int* __restrict__ seg, float* __restrict__ dst,
                    const float* __restrict__ resid, int nrows, float invR, int direct)
{
    size_t gid = (size_t)blockIdx.x * 256 + threadIdx.x;
    size_t total = (size_t)nrows * D;
    if (gid >= total) return;
    int c = (int)(gid & (D - 1));

    float mc = stats[c] * invR;
    float vc = stats[D + c] * invR - mc * mc;
    float mg = stats[2 * D + c] * invR;
    float vg = stats[3 * D + c] * invR - mg * mg;

    float xc = (core[gid] - mc) * rsqrtf(vc + 1e-5f) * gc[c] + bc[c];
    float xg = (gate[gid] - mg) * rsqrtf(vg + 1e-5f) * gg[c] + bg[c];
    float sc = 1.f / (1.f + __expf(-xc));
    float sg = 1.f / (1.f + __expf(-xg));
    float m  = (xc * sc) * sg;

    if (direct) {
        dst[gid] = m + resid[gid];
    } else {
        size_t row = gid >> 6;
        atomicAdd(&dst[(size_t)seg[row] * D + c], m);
    }
}

// ---------------------------------------------------------------------------
// Pass 3: out = acc @ W (64x64, fp32 WMMA 16x16x4) + residual.
// Block = 128 threads (4 waves), wave = col tile.
// ---------------------------------------------------------------------------
__global__ __launch_bounds__(128)
void chgnet_out_gemm(const float* __restrict__ acc, const float* __restrict__ W,
                     const float* __restrict__ resid, float* __restrict__ out, int nrows)
{
    __shared__ float ldsA[16 * D];
    const int tid  = threadIdx.x;
    const int ct   = tid >> 5;
    const int lane = tid & 31;
    const int half = lane >> 4;
    const int ln   = lane & 15;
    const int col  = ct * 16 + ln;

    // B 4x16 f32 layout: VGPR v, lanes 0-15: K=v ; lanes 16-31: K=v+2.
    v2f bfrag[16];
#pragma unroll
    for (int ks = 0; ks < 16; ++ks) {
        int K = ks * 4 + half * 2;
        bfrag[ks][0] = W[(K)     * D + col];
        bfrag[ks][1] = W[(K + 1) * D + col];
    }

    const int ntiles = nrows >> 4;
    for (int tile = blockIdx.x; tile < ntiles; tile += gridDim.x) {
        const int rbase = tile << 4;
        __syncthreads();
        for (int idx = tid; idx < 16 * D; idx += 128)
            ldsA[idx] = acc[(size_t)rbase * D + idx];
        __syncthreads();

        v8f c = {};
#pragma unroll
        for (int ks = 0; ks < 16; ++ks) {
            // A 16x4 f32 layout: VGPR0 K=0(lo)/2(hi), VGPR1 K=1(lo)/3(hi)
            v2f a;
            int K = ks * 4 + half * 2;
            a[0] = ldsA[ln * D + K];
            a[1] = ldsA[ln * D + K + 1];
            c = __builtin_amdgcn_wmma_f32_16x16x4_f32(
                false, a, false, bfrag[ks], (short)0, c, false, false);
        }
#pragma unroll
        for (int r = 0; r < 8; ++r) {
            int row = rbase + r + half * 8;
            out[(size_t)row * D + col] = c[r] + resid[(size_t)row * D + col];
        }
    }
}

// ---------------------------------------------------------------------------
extern "C" void kernel_launch(void* const* d_in, const int* in_sizes, int n_in,
                              void* d_out, int out_size, void* d_ws, size_t ws_size,
                              hipStream_t stream) {
    const float* vertex = (const float*)d_in[0];
    const float* edge   = (const float*)d_in[1];
    const float* angle  = (const float*)d_in[2];
    const int*   eindex = (const int*)d_in[3];
    const int*   kidx   = (const int*)d_in[4];
    const int*   iidx   = (const int*)d_in[5];
    const int*   jidx   = (const int*)d_in[6];
    const float* WcA = (const float*)d_in[7];
    const float* WgA = (const float*)d_in[8];
    const float* WoA = (const float*)d_in[9];
    const float* WcB = (const float*)d_in[10];
    const float* WgB = (const float*)d_in[11];
    const float* WoB = (const float*)d_in[12];
    const float* WcN = (const float*)d_in[13];
    const float* WgN = (const float*)d_in[14];
    const float* g_ac = (const float*)d_in[15];
    const float* g_ag = (const float*)d_in[16];
    const float* g_bc = (const float*)d_in[17];
    const float* g_bg = (const float*)d_in[18];
    const float* g_nc = (const float*)d_in[19];
    const float* g_ng = (const float*)d_in[20];
    const float* b_ac = (const float*)d_in[21];
    const float* b_ag = (const float*)d_in[22];
    const float* b_bc = (const float*)d_in[23];
    const float* b_bg = (const float*)d_in[24];
    const float* b_nc = (const float*)d_in[25];
    const float* b_ng = (const float*)d_in[26];

    const int N = in_sizes[0] / D;
    const int E = in_sizes[1] / D;
    const int T = in_sizes[2] / D;
    const int* src = eindex;
    const int* dst = eindex + E;

    float* out   = (float*)d_out;
    float* v_new = out;
    float* e_new = out + (size_t)N * D;
    float* a_new = out + (size_t)(N + E) * D;

    // Workspace layout
    char* ws = (char*)d_ws;
    size_t off = 0;
    auto walloc = [&](size_t bytes) -> void* {
        void* p = ws + off;
        off = (off + bytes + 255) & ~(size_t)255;
        return p;
    };
    float* statsA  = (float*)walloc(4 * D * sizeof(float));
    float* statsB  = (float*)walloc(4 * D * sizeof(float));
    float* statsN  = (float*)walloc(4 * D * sizeof(float));
    float* v_accum = (float*)walloc((size_t)N * D * sizeof(float));
    float* e_accum = (float*)walloc((size_t)E * D * sizeof(float));
    size_t maxRows = (size_t)(E > T ? E : T);
    float* coreBuf = (float*)walloc(maxRows * D * sizeof(float));
    float* gateBuf = (float*)walloc(maxRows * D * sizeof(float));

    // Zero accumulators / stats (graph-capture-safe, re-executed each replay)
    hipMemsetAsync(statsA, 0, 3 * 4 * D * sizeof(float) + 2 * 256, stream);
    hipMemsetAsync(v_accum, 0, (size_t)N * D * sizeof(float), stream);
    hipMemsetAsync(e_accum, 0, (size_t)E * D * sizeof(float), stream);

    const int GBLK = 1024;
    auto grid2 = [&](int rows) { return (int)(((size_t)rows * D + 255) / 256); };
    auto grid3 = [&](int rows) { int t = rows >> 4; return t < 8192 ? t : 8192; };

    // ---- AtomConv: msg = [vertex[src], edge, vertex[dst]] (K=192) ----
    chgnet_gemm_stats<192><<<GBLK, 256, 0, stream>>>(
        vertex, src, edge, nullptr, vertex, dst, nullptr,
        WcA, WgA, coreBuf, gateBuf, statsA, E);
    chgnet_bn_gate<<<grid2(E), 256, 0, stream>>>(
        coreBuf, gateBuf, statsA, g_ac, b_ac, g_ag, b_ag,
        src, v_accum, nullptr, E, 1.f / (float)E, 0);
    chgnet_out_gemm<<<grid3(N), 128, 0, stream>>>(v_accum, WoA, vertex, v_new, N);

    // ---- BondConv: tot = [v_new[j], edge[k], edge[i], angle] (K=256) ----
    chgnet_gemm_stats<256><<<GBLK, 256, 0, stream>>>(
        v_new, jidx, edge, kidx, edge, iidx, angle,
        WcB, WgB, coreBuf, gateBuf, statsB, T);
    chgnet_bn_gate<<<grid2(T), 256, 0, stream>>>(
        coreBuf, gateBuf, statsB, g_bc, b_bc, g_bg, b_bg,
        kidx, e_accum, nullptr, T, 1.f / (float)T, 0);
    chgnet_out_gemm<<<grid3(E), 128, 0, stream>>>(e_accum, WoB, edge, e_new, E);

    // ---- AngleConv: tot2 = [v_new[j], e_new[k], e_new[i], angle] (K=256) ----
    chgnet_gemm_stats<256><<<GBLK, 256, 0, stream>>>(
        v_new, jidx, e_new, kidx, e_new, iidx, angle,
        WcN, WgN, coreBuf, gateBuf, statsN, T);
    chgnet_bn_gate<<<grid2(T), 256, 0, stream>>>(
        coreBuf, gateBuf, statsN, g_nc, b_nc, g_ng, b_ng,
        nullptr, a_new, angle, T, 1.f / (float)T, 1);
}
